// Trans_Global_SSG_v2_33655363732082
// MI455X (gfx1250) — matrix operand order repaired and checked by reading, not compiled, not run on hardware
//
#include <hip/hip_runtime.h>

typedef __attribute__((ext_vector_type(16))) _Float16 v16h;
typedef __attribute__((ext_vector_type(8)))  float    v8f;

#define BB      4
#define NN      4096
#define NPOINT  1024
#define NSAMPLE 32
#define DIN     64
#define CC      128

// workspace offsets (bytes)
#define OFF_FPSIDX 0u
#define OFF_GIDX   16384u
#define OFF_NXYZ   540672u
#define OFF_FEAT   1048576u                     // half[4*32*1024*128] = 32MB
#define OFF_Q      (OFF_FEAT + 33554432u)
#define OFF_K      (OFF_Q   + 33554432u)
#define OFF_V      (OFF_K   + 33554432u)
#define OFF_ATT    (OFF_V   + 33554432u)
#define OFF_W0P    (OFF_ATT + 33554432u)
#define OFF_W1P    (OFF_W0P + 65536u)
#define OFF_W2P    (OFF_W1P + 65536u)
#define OFF_WQP    (OFF_W2P + 65536u)
#define OFF_WKP    (OFF_WQP + 65536u)
#define OFF_WVP    (OFF_WKP + 65536u)
#define OFF_WOP    (OFF_WVP + 65536u)
#define OFF_WPOSP  (OFF_WOP + 65536u)
#define OFF_PTST   (OFF_WPOSP + 65536u)         // half[4*4096*64] = 2MB

__device__ __forceinline__ v8f wmma16(v16h a, v16h b, v8f c) {
  return __builtin_amdgcn_wmma_f32_16x16x32_f16(false, a, false, b, (short)0, c, false, false);
}

// A fragment (16x32 f16) from a row-major f16 row: k = 16h + {0..15}
__device__ __forceinline__ v16h ldA16(const _Float16* p) {
  union { uint4 u[2]; v16h v; } t;
  t.u[0] = *(const uint4*)(p);
  t.u[1] = *(const uint4*)(p + 8);
  return t.v;
}
// B fragment: 16 contiguous halves (pre-swizzled weights / LDS rows)
__device__ __forceinline__ v16h ldB16(const _Float16* p) {
  union { uint4 u[2]; v16h v; } t;
  t.u[0] = *(const uint4*)(p);
  t.u[1] = *(const uint4*)(p + 8);
  return t.v;
}

// ---------------------------------------------------------------- weight pre-swizzle -> f16 B-fragment order
// out[((nt*nkc + kc)*32 + lane)*16 + e] = W(k = kc*32 + 16*(lane/16) + e, n = nt*16 + lane%16)
// mode 1: W[k*ncols + n]          (wq,wk,wv,wo)
// mode 0: W[n*kreal + k]          (w1,w2,wpos)
// mode 2: permuted layer-1 order  (w0: k<64 -> cin=k+3 (points), 64..66 -> cin=k-64 (xyz), else 0)
__global__ void pack_b_kernel(const float* __restrict__ W, _Float16* __restrict__ out,
                              int nkc, int nnt, int kreal, int ncols, int mode) {
  int idx = blockIdx.x * 256 + threadIdx.x;
  int total = nkc * nnt * 512;
  if (idx >= total) return;
  int e = idx & 15;
  int lane = (idx >> 4) & 31;
  int frag = idx >> 9;
  int kc = frag % nkc, nt = frag / nkc;
  int h = lane >> 4, r = lane & 15;
  int k = kc * 32 + (h << 4) + e;
  int n = nt * 16 + r;
  float v = 0.f;
  if (mode == 2) {
    int cin = (k < 64) ? (k + 3) : ((k < 67) ? (k - 64) : -1);
    if (cin >= 0) v = W[n * 67 + cin];
  } else if (k < kreal) {
    v = mode ? W[k * ncols + n] : W[n * kreal + k];
  }
  out[idx] = (_Float16)v;
}

// ---------------------------------------------------------------- points transpose: [b][c][n] f32 -> [b][n][c] f16
__global__ void transpose_points_kernel(const float* __restrict__ points, _Float16* __restrict__ ptsT) {
  int idx = blockIdx.x * 256 + threadIdx.x;      // b,c,n order: coalesced reads
  if (idx >= BB * DIN * NN) return;
  int n = idx & (NN - 1);
  int c = (idx >> 12) & (DIN - 1);
  int b = idx >> 18;
  ptsT[((size_t)b * NN + n) * DIN + c] = (_Float16)points[idx];
}

// ---------------------------------------------------------------- FPS
__global__ void __launch_bounds__(1024) fps_kernel(const float* __restrict__ xyz,
                                                   int* __restrict__ fpsIdx,
                                                   float* __restrict__ nXyz,
                                                   float* __restrict__ out0) {
  __shared__ float scx, scy, scz;
  __shared__ int   sfar;
  __shared__ float sval[32];
  __shared__ int   sidx[32];
  const int b = blockIdx.x, tid = threadIdx.x, wv = tid >> 5, lane = tid & 31;
  const float* Xb = xyz + b * 3 * NN;
  float dist[4] = {1e10f, 1e10f, 1e10f, 1e10f};
  if (tid == 0) sfar = 0;
  __syncthreads();
  for (int it = 0; it < NPOINT; ++it) {
    if (tid == 0) {
      int f = sfar;
      float x = Xb[f], y = Xb[NN + f], z = Xb[2 * NN + f];
      scx = x; scy = y; scz = z;
      fpsIdx[b * NPOINT + it] = f;
      float* nz = nXyz + (b * NPOINT + it) * 3;
      nz[0] = x; nz[1] = y; nz[2] = z;
      out0[b * 3 * NPOINT + it] = x;
      out0[b * 3 * NPOINT + NPOINT + it] = y;
      out0[b * 3 * NPOINT + 2 * NPOINT + it] = z;
    }
    __syncthreads();
    float cx = scx, cy = scy, cz = scz;
    float best = -1.0f; int bestn = 0;
#pragma unroll
    for (int j = 0; j < 4; ++j) {
      int n = j * 1024 + tid;
      float dx = Xb[n] - cx, dy = Xb[NN + n] - cy, dz = Xb[2 * NN + n] - cz;
      float d = dx * dx + dy * dy + dz * dz;
      d = fminf(dist[j], d);
      dist[j] = d;
      if (d > best) { best = d; bestn = n; }
    }
#pragma unroll
    for (int o = 1; o < 32; o <<= 1) {
      float ob = __shfl_xor(best, o);
      int   on = __shfl_xor(bestn, o);
      if (ob > best || (ob == best && on < bestn)) { best = ob; bestn = on; }
    }
    if (lane == 0) { sval[wv] = best; sidx[wv] = bestn; }
    __syncthreads();
    if (wv == 0) {
      best = sval[lane]; bestn = sidx[lane];
#pragma unroll
      for (int o = 1; o < 32; o <<= 1) {
        float ob = __shfl_xor(best, o);
        int   on = __shfl_xor(bestn, o);
        if (ob > best || (ob == best && on < bestn)) { best = ob; bestn = on; }
      }
      if (lane == 0) sfar = bestn;
    }
    __syncthreads();
  }
}

// ---------------------------------------------------------------- ball query (one wave / group)
__global__ void ballquery_kernel(const float* __restrict__ xyz,
                                 const float* __restrict__ nXyz,
                                 int* __restrict__ gIdx) {
  const int tid = threadIdx.x, wv = tid >> 5, lane = tid & 31;
  const int g = blockIdx.x * 8 + wv;           // 0..4095
  const int b = g >> 10;
  const float* Xb = xyz + b * 3 * NN;
  const float* cp = nXyz + g * 3;
  const float cx = cp[0], cy = cp[1], cz = cp[2];
  const float r2 = 0.2f * 0.2f;
  int* out = gIdx + g * NSAMPLE;
  int count = 0;
  for (int base = 0; base < NN && count < NSAMPLE; base += 32) {
    int n = base + lane;
    float dx = Xb[n] - cx, dy = Xb[NN + n] - cy, dz = Xb[2 * NN + n] - cz;
    bool inr = (dx * dx + dy * dy + dz * dz) <= r2;
    unsigned msk = (unsigned)__ballot(inr);
    if (inr) {
      int pos = count + __popc(msk & ((1u << lane) - 1u));
      if (pos < NSAMPLE) out[pos] = n;
    }
    count += __popc(msk);
  }
  if (count < NSAMPLE) {
    int first = 0;
    if (lane == 0) first = out[0];     // same-wave store->load kept in order
    first = __shfl(first, 0);
    int p = count + lane;
    if (p < NSAMPLE) out[p] = first;
  }
}

// ---------------------------------------------------------------- grouping + 3-layer MLP + pos-enc (WMMA)
__global__ void group_mlp_kernel(const float* __restrict__ xyz, const _Float16* __restrict__ ptsT,
                                 const _Float16* __restrict__ w0p, const float* __restrict__ b0,
                                 const _Float16* __restrict__ w1p, const float* __restrict__ b1,
                                 const _Float16* __restrict__ w2p, const float* __restrict__ b2,
                                 const _Float16* __restrict__ wposp,
                                 const int* __restrict__ gIdx, const float* __restrict__ nXyz,
                                 _Float16* __restrict__ feat) {
  __shared__ _Float16 sact0[8][16][72];
  __shared__ _Float16 sact1[8][16][72];
  const int tid = threadIdx.x, wv = tid >> 5, lane = tid & 31;
  const int h = lane >> 4, r = lane & 15;
  const int tile = blockIdx.x * 8 + wv;      // 0..8191
  const int b = tile >> 11;
  const int rem = tile & 2047;
  const int s = rem >> 1;
  const int half = rem & 1;
  const int ks = half * 16 + r;
  const int gbase = (b * NPOINT + s) * NSAMPLE;
  const int gi = gIdx[gbase + ks];
  const float* Xb = xyz + b * 3 * NN;
  const float nx = Xb[gi], ny = Xb[NN + gi], nz = Xb[2 * NN + gi];
  const float* cen = nXyz + (b * NPOINT + s) * 3;
  const float fx = nx - cen[0], fy = ny - cen[1], fz = nz - cen[2];
  const _Float16* prow = ptsT + ((size_t)b * NN + gi) * DIN;
  const v8f z8 = {0.f, 0.f, 0.f, 0.f, 0.f, 0.f, 0.f, 0.f};

  // ---- layer 1: permuted K order [points(64), xyz-norm(3), pad] -> 64
  v8f acc1[4];
#pragma unroll
  for (int nt = 0; nt < 4; ++nt) acc1[nt] = z8;
#pragma unroll
  for (int kc = 0; kc < 2; ++kc) {
    v16h A = ldA16(prow + kc * 32 + 16 * h);
#pragma unroll
    for (int nt = 0; nt < 4; ++nt)
      acc1[nt] = wmma16(A, ldB16(w0p + ((nt * 3 + kc) * 32 + lane) * 16), acc1[nt]);
  }
  {
    v16h A;
#pragma unroll
    for (int e = 0; e < 16; ++e) A[e] = (_Float16)0.f;
    if (h == 0) { A[0] = (_Float16)fx; A[1] = (_Float16)fy; A[2] = (_Float16)fz; }
#pragma unroll
    for (int nt = 0; nt < 4; ++nt)
      acc1[nt] = wmma16(A, ldB16(w0p + ((nt * 3 + 2) * 32 + lane) * 16), acc1[nt]);
  }
#pragma unroll
  for (int nt = 0; nt < 4; ++nt) {
    float bias = b0[nt * 16 + r];
#pragma unroll
    for (int v = 0; v < 8; ++v)
      sact0[wv][v + 8 * h][nt * 16 + r] = (_Float16)fmaxf(acc1[nt][v] + bias, 0.f);
  }
  asm volatile("s_wait_dscnt 0" ::: "memory");

  // ---- layer 2: 64 -> 64
  v8f acc2[4];
#pragma unroll
  for (int nt = 0; nt < 4; ++nt) acc2[nt] = z8;
#pragma unroll
  for (int kc = 0; kc < 2; ++kc) {
    v16h A = ldA16(&sact0[wv][r][kc * 32 + 16 * h]);
#pragma unroll
    for (int nt = 0; nt < 4; ++nt)
      acc2[nt] = wmma16(A, ldB16(w1p + ((nt * 2 + kc) * 32 + lane) * 16), acc2[nt]);
  }
#pragma unroll
  for (int nt = 0; nt < 4; ++nt) {
    float bias = b1[nt * 16 + r];
#pragma unroll
    for (int v = 0; v < 8; ++v)
      sact1[wv][v + 8 * h][nt * 16 + r] = (_Float16)fmaxf(acc2[nt][v] + bias, 0.f);
  }
  asm volatile("s_wait_dscnt 0" ::: "memory");

  // ---- layer 3: 64 -> 128
  v8f acc3[8];
#pragma unroll
  for (int nt = 0; nt < 8; ++nt) acc3[nt] = z8;
#pragma unroll
  for (int kc = 0; kc < 2; ++kc) {
    v16h A = ldA16(&sact1[wv][r][kc * 32 + 16 * h]);
#pragma unroll
    for (int nt = 0; nt < 8; ++nt)
      acc3[nt] = wmma16(A, ldB16(w2p + ((nt * 2 + kc) * 32 + lane) * 16), acc3[nt]);
  }

  // ---- positional encoding as WMMA: penc = pos(16x3 pad 32) @ wpos^T
  v8f accP[8];
#pragma unroll
  for (int nt = 0; nt < 8; ++nt) accP[nt] = z8;
  {
    v16h A;
#pragma unroll
    for (int e = 0; e < 16; ++e) A[e] = (_Float16)0.f;
    if (h == 0) { A[0] = (_Float16)nx; A[1] = (_Float16)ny; A[2] = (_Float16)nz; }
#pragma unroll
    for (int nt = 0; nt < 8; ++nt)
      accP[nt] = wmma16(A, ldB16(wposp + (nt * 32 + lane) * 16), accP[nt]);
  }

  // ---- epilogue: relu(mlp + bias) + penc, store p (f16)
#pragma unroll
  for (int nt = 0; nt < 8; ++nt) {
    int c = nt * 16 + r;
    float bias = b2[c];
#pragma unroll
    for (int v = 0; v < 8; ++v) {
      int kk = half * 16 + v + 8 * h;
      float val = fmaxf(acc3[nt][v] + bias, 0.f) + accP[nt][v];
      feat[(((size_t)(b * 32 + kk)) * NPOINT + s) * CC + c] = (_Float16)val;
    }
  }
}

// ---------------------------------------------------------------- Q/K/V projections (WMMA)
__global__ void qkv_kernel(const _Float16* __restrict__ feat,
                           const _Float16* __restrict__ wqp, const _Float16* __restrict__ wkp,
                           const _Float16* __restrict__ wvp,
                           _Float16* __restrict__ q, _Float16* __restrict__ k,
                           _Float16* __restrict__ v) {
  const int tid = threadIdx.x, wv = tid >> 5, lane = tid & 31;
  const int h = lane >> 4, r = lane & 15;
  const int gw = blockIdx.x * 8 + wv;         // 0..24575
  const int which = gw >> 13;
  const int mt = gw & 8191;
  const _Float16* Wp = (which == 0) ? wqp : (which == 1) ? wkp : wvp;
  _Float16* O = (which == 0) ? q : (which == 1) ? k : v;
  const size_t rowbase = (size_t)mt * 16;
  v16h A[4];
#pragma unroll
  for (int kc = 0; kc < 4; ++kc)
    A[kc] = ldA16(feat + (rowbase + r) * CC + kc * 32 + 16 * h);
  const v8f z8 = {0.f, 0.f, 0.f, 0.f, 0.f, 0.f, 0.f, 0.f};
  v8f acc[8];
#pragma unroll
  for (int nt = 0; nt < 8; ++nt) acc[nt] = z8;
#pragma unroll
  for (int nt = 0; nt < 8; ++nt)
#pragma unroll
    for (int kc = 0; kc < 4; ++kc)
      acc[nt] = wmma16(A[kc], ldB16(Wp + ((nt * 4 + kc) * 32 + lane) * 16), acc[nt]);
#pragma unroll
  for (int nt = 0; nt < 8; ++nt)
#pragma unroll
    for (int vv = 0; vv < 8; ++vv)
      O[(rowbase + vv + 8 * h) * CC + nt * 16 + r] = (_Float16)acc[nt][vv];
}

// ---------------------------------------------------------------- flash attention (WMMA, online softmax)
#define SKSTR 136
#define SVSTR 40
__global__ void attn_kernel(const _Float16* __restrict__ q, const _Float16* __restrict__ k,
                            const _Float16* __restrict__ v, _Float16* __restrict__ att) {
  __shared__ _Float16 sK[32][SKSTR];        // key-major (rows = key, cols = c)
  __shared__ _Float16 sVt[128][SVSTR];      // transposed (rows = c, cols = key)
  __shared__ _Float16 sP[8][16][40];
  const int tid = threadIdx.x, wv = tid >> 5, lane = tid & 31;
  const int h = lane >> 4, r = lane & 15;
  const int bk = blockIdx.x >> 3;             // head index 0..127
  const int qt = blockIdx.x & 7;
  const size_t hb = (size_t)bk * NPOINT * CC;
  const int qrow = qt * 128 + wv * 16;
  const float scale = 0.08838834764831845f;   // 1/sqrt(128)

  v16h Af[4];
#pragma unroll
  for (int kc = 0; kc < 4; ++kc)
    Af[kc] = ldA16(q + hb + (size_t)(qrow + r) * CC + kc * 32 + 16 * h);

  const v8f z8 = {0.f, 0.f, 0.f, 0.f, 0.f, 0.f, 0.f, 0.f};
  v8f O[8];
  float rm[8], rl[8];
#pragma unroll
  for (int i = 0; i < 8; ++i) { O[i] = z8; rm[i] = -3e38f; rl[i] = 0.f; }

  for (int kt = 0; kt < 32; ++kt) {
    __syncthreads();
    // K tile: CDNA5 async global->LDS copy (no VGPR round-trip, tracked by ASYNCcnt)
    // V tile: staged through VGPRs with transpose into sVt
    for (int i = tid; i < 512; i += 256) {
      int row = i >> 4, seg = i & 15;
      unsigned lk = (unsigned)(uintptr_t)&sK[row][seg * 8];
      unsigned long long ga =
          (unsigned long long)(uintptr_t)&k[hb + (size_t)(kt * 32 + row) * CC + seg * 8];
      asm volatile("global_load_async_to_lds_b128 %0, %1, off"
                   :: "v"(lk), "v"(ga) : "memory");
      union { uint4 u; _Float16 hh[8]; } tv;
      tv.u = *(const uint4*)&v[hb + (size_t)(kt * 32 + row) * CC + seg * 8];
#pragma unroll
      for (int j = 0; j < 8; ++j) sVt[seg * 8 + j][row] = tv.hh[j];
    }
    // prefetch next tiles into cache while this one is consumed
    if (kt + 1 < 32 && tid < 128) {
      const char* pk = (const char*)&k[hb + (size_t)((kt + 1) * 32) * CC];
      const char* pv = (const char*)&v[hb + (size_t)((kt + 1) * 32) * CC];
      __builtin_prefetch(pk + tid * 64, 0, 0);
      __builtin_prefetch(pv + tid * 64, 0, 0);
    }
    asm volatile("s_wait_asynccnt 0" ::: "memory");
    __syncthreads();

    // S = Q K^T for a 16x32 slab (two 16x16 N-tiles)
    v8f accS[2];
    accS[0] = z8; accS[1] = z8;
#pragma unroll
    for (int jt = 0; jt < 2; ++jt)
#pragma unroll
      for (int kc = 0; kc < 4; ++kc)
        accS[jt] = wmma16(Af[kc], ldB16(&sK[jt * 16 + r][kc * 32 + 16 * h]), accS[jt]);

    // online softmax
    float f[8];
#pragma unroll
    for (int vv = 0; vv < 8; ++vv) {
      float s0 = accS[0][vv] * scale, s1 = accS[1][vv] * scale;
      float mx = fmaxf(s0, s1);
#pragma unroll
      for (int o = 1; o < 16; o <<= 1) mx = fmaxf(mx, __shfl_xor(mx, o));
      float mnew = fmaxf(rm[vv], mx);
      float p0 = __expf(s0 - mnew), p1 = __expf(s1 - mnew);
      float rs = p0 + p1;
#pragma unroll
      for (int o = 1; o < 16; o <<= 1) rs += __shfl_xor(rs, o);
      f[vv] = __expf(rm[vv] - mnew);
      rl[vv] = rl[vv] * f[vv] + rs;
      rm[vv] = mnew;
      int m = vv + 8 * h;
      sP[wv][m][r] = (_Float16)p0;
      sP[wv][m][16 + r] = (_Float16)p1;
    }
#pragma unroll
    for (int nt = 0; nt < 8; ++nt)
#pragma unroll
      for (int vv = 0; vv < 8; ++vv) O[nt][vv] *= f[vv];
    asm volatile("s_wait_dscnt 0" ::: "memory");

    // O += P V
    v16h Pf = ldA16(&sP[wv][r][16 * h]);
#pragma unroll
    for (int nt = 0; nt < 8; ++nt)
      O[nt] = wmma16(Pf, ldB16(&sVt[nt * 16 + r][16 * h]), O[nt]);
  }

#pragma unroll
  for (int nt = 0; nt < 8; ++nt)
#pragma unroll
    for (int vv = 0; vv < 8; ++vv)
      att[hb + (size_t)(qrow + vv + 8 * h) * CC + nt * 16 + r] = (_Float16)(O[nt][vv] / rl[vv]);
}

// ---------------------------------------------------------------- (attnV)@wo + residual + max over samples
__global__ void outproj_kernel(const _Float16* __restrict__ att, const _Float16* __restrict__ feat,
                               const _Float16* __restrict__ wop, float* __restrict__ out1) {
  const int tid = threadIdx.x, wv = tid >> 5, lane = tid & 31;
  const int h = lane >> 4, r = lane & 15;
  const int gw = blockIdx.x * 8 + wv;          // 0..32767
  const int nt = gw & 7;
  const int bs = gw >> 3;
  const int b = bs >> 10, s = bs & 1023;
  const v8f z8 = {0.f, 0.f, 0.f, 0.f, 0.f, 0.f, 0.f, 0.f};
  float mx = -3e38f;
#pragma unroll
  for (int kt2 = 0; kt2 < 2; ++kt2) {
    v8f acc = z8;
#pragma unroll
    for (int kc = 0; kc < 4; ++kc) {
      v16h A = ldA16(att + ((size_t)(b * 32 + kt2 * 16 + r) * NPOINT + s) * CC + kc * 32 + 16 * h);
      acc = wmma16(A, ldB16(wop + ((nt * 4 + kc) * 32 + lane) * 16), acc);
    }
#pragma unroll
    for (int vv = 0; vv < 8; ++vv) {
      int kk = kt2 * 16 + vv + 8 * h;
      float pv = (float)feat[((size_t)(b * 32 + kk) * NPOINT + s) * CC + nt * 16 + r];
      mx = fmaxf(mx, acc[vv] + pv);
    }
  }
  mx = fmaxf(mx, __shfl_xor(mx, 16));
  if (h == 0)
    out1[(size_t)b * CC * NPOINT + (nt * 16 + r) * NPOINT + s] = mx;
}

// ---------------------------------------------------------------- launch
extern "C" void kernel_launch(void* const* d_in, const int* in_sizes, int n_in,
                              void* d_out, int out_size, void* d_ws, size_t ws_size,
                              hipStream_t stream) {
  (void)in_sizes; (void)n_in; (void)out_size; (void)ws_size;
  const float* xyz    = (const float*)d_in[0];
  const float* points = (const float*)d_in[1];
  const float* w0 = (const float*)d_in[2];
  const float* b0 = (const float*)d_in[3];
  const float* w1 = (const float*)d_in[4];
  const float* b1 = (const float*)d_in[5];
  const float* w2 = (const float*)d_in[6];
  const float* b2 = (const float*)d_in[7];
  const float* wq = (const float*)d_in[8];
  const float* wk = (const float*)d_in[9];
  const float* wvw = (const float*)d_in[10];
  const float* wo = (const float*)d_in[11];
  const float* wpos = (const float*)d_in[12];
  float* out = (float*)d_out;
  char* ws = (char*)d_ws;

  int*      fpsIdx = (int*)(ws + OFF_FPSIDX);
  int*      gIdx   = (int*)(ws + OFF_GIDX);
  float*    nXyz   = (float*)(ws + OFF_NXYZ);
  _Float16* feat   = (_Float16*)(ws + OFF_FEAT);
  _Float16* qh     = (_Float16*)(ws + OFF_Q);
  _Float16* kh     = (_Float16*)(ws + OFF_K);
  _Float16* vh     = (_Float16*)(ws + OFF_V);
  _Float16* att    = (_Float16*)(ws + OFF_ATT);
  _Float16* w0p    = (_Float16*)(ws + OFF_W0P);
  _Float16* w1p    = (_Float16*)(ws + OFF_W1P);
  _Float16* w2p    = (_Float16*)(ws + OFF_W2P);
  _Float16* wqp    = (_Float16*)(ws + OFF_WQP);
  _Float16* wkp    = (_Float16*)(ws + OFF_WKP);
  _Float16* wvp    = (_Float16*)(ws + OFF_WVP);
  _Float16* wop    = (_Float16*)(ws + OFF_WOP);
  _Float16* wposp  = (_Float16*)(ws + OFF_WPOSP);
  _Float16* ptsT   = (_Float16*)(ws + OFF_PTST);

  // pre-swizzle weights into WMMA B-fragment order (f16)
  pack_b_kernel<<<24, 256, 0, stream>>>(w0, w0p, 3, 4, 96, 64, 2);
  pack_b_kernel<<<16, 256, 0, stream>>>(w1, w1p, 2, 4, 64, 64, 0);
  pack_b_kernel<<<32, 256, 0, stream>>>(w2, w2p, 2, 8, 64, 128, 0);
  pack_b_kernel<<<64, 256, 0, stream>>>(wq, wqp, 4, 8, 128, 128, 1);
  pack_b_kernel<<<64, 256, 0, stream>>>(wk, wkp, 4, 8, 128, 128, 1);
  pack_b_kernel<<<64, 256, 0, stream>>>(wvw, wvp, 4, 8, 128, 128, 1);
  pack_b_kernel<<<64, 256, 0, stream>>>(wo, wop, 4, 8, 128, 128, 1);
  pack_b_kernel<<<16, 256, 0, stream>>>(wpos, wposp, 1, 8, 3, 128, 0);
  transpose_points_kernel<<<4096, 256, 0, stream>>>(points, ptsT);

  fps_kernel<<<BB, 1024, 0, stream>>>(xyz, fpsIdx, nXyz, out);
  ballquery_kernel<<<512, 256, 0, stream>>>(xyz, nXyz, gIdx);
  group_mlp_kernel<<<1024, 256, 0, stream>>>(xyz, ptsT, w0p, b0, w1p, b1, w2p, b2, wposp,
                                             gIdx, nXyz, feat);
  qkv_kernel<<<3072, 256, 0, stream>>>(feat, wqp, wkp, wvp, qh, kh, vh);
  attn_kernel<<<1024, 256, 0, stream>>>(qh, kh, vh, att);
  outproj_kernel<<<4096, 256, 0, stream>>>(att, feat, wop, out + BB * 3 * NPOINT);
}
